// GNNReID_31619549233289
// MI455X (gfx1250) — matrix operand order repaired and checked by reading, not compiled, not run on hardware
//
#include <hip/hip_runtime.h>
#include <hip/hip_bf16.h>

// Problem constants (from reference)
#define NN   4096      // nodes
#define EE   131072    // edges
#define DD   2048      // feature dim
#define HH   4         // heads
#define HD   512       // head dim
#define LL   2         // layers

typedef __attribute__((ext_vector_type(16))) __bf16 v16bf;
typedef __attribute__((ext_vector_type(8)))  float  v8f;

// ---------------- helpers ----------------

__device__ __forceinline__ __bf16 f2bf(float f) {
    unsigned u = __float_as_uint(f);
    unsigned r = (u + 0x7FFFu + ((u >> 16) & 1u)) >> 16;   // RNE
    unsigned short s = (unsigned short)r;
    __bf16 b;
    __builtin_memcpy(&b, &s, 2);
    return b;
}

// order-preserving float -> uint encoding for atomicMax
__device__ __forceinline__ unsigned f2ord(float f) {
    unsigned b = __float_as_uint(f);
    return (b & 0x80000000u) ? ~b : (b | 0x80000000u);
}
__device__ __forceinline__ float ord2f(unsigned e) {
    unsigned b = (e & 0x80000000u) ? (e & 0x7FFFFFFFu) : ~e;
    return __uint_as_float(b);
}

// ---------------- small utility kernels ----------------

__global__ void zero16_kernel(uint4* __restrict__ p, size_t n16) {
    size_t i = (size_t)blockIdx.x * blockDim.x + threadIdx.x;
    size_t stride = (size_t)gridDim.x * blockDim.x;
    uint4 z = {0u, 0u, 0u, 0u};
    for (; i < n16; i += stride) p[i] = z;
}

__global__ void f32_to_bf16_kernel(const float* __restrict__ src,
                                   __bf16* __restrict__ dst, size_t n) {
    size_t i = (size_t)blockIdx.x * blockDim.x + threadIdx.x;
    size_t stride = (size_t)gridDim.x * blockDim.x;
    for (; i < n; i += stride) dst[i] = f2bf(src[i]);
}

// Tiled transpose + convert: dst[l][j][k] = bf16(src[l][k][j]); D x D per layer.
__global__ __launch_bounds__(256)
void transpose_f32_to_bf16_kernel(const float* __restrict__ src, __bf16* __restrict__ dst) {
    __shared__ float tile[32][33];
    const int tx = threadIdx.x & 31;
    const int ty = threadIdx.x >> 5;          // 0..7
    const int bx = blockIdx.x * 32;           // src col
    const int by = blockIdx.y * 32;           // src row
    const float* s = src + (size_t)blockIdx.z * DD * DD;
    __bf16*      d = dst + (size_t)blockIdx.z * DD * DD;
    #pragma unroll
    for (int i = 0; i < 4; ++i)
        tile[ty + i * 8][tx] = s[(size_t)(by + ty + i * 8) * DD + bx + tx];
    __syncthreads();
    #pragma unroll
    for (int i = 0; i < 4; ++i)
        d[(size_t)(bx + ty + i * 8) * DD + by + tx] = f2bf(tile[tx][ty + i * 8]);
}

// ---------------- WMMA GEMM: C[M,Nc] = A[M,K](bf16) @ Bt[Nc,K](bf16)^T + bias ----------------
// Both operands are K-contiguous, so BOTH tiles stream memory->LDS via
// GLOBAL_LOAD_ASYNC_TO_LDS_B128 (ASYNCcnt), double-buffered so tile i+1's loads overlap
// tile i's 8 WMMAs per wave. Steady-state loop is branch-free: the final two tiles are
// peeled so every in-loop wait is s_wait_asynccnt 0x4. Block tile 128x128, wave tile 32x64.

#define TM 128
#define TN 128
#define TK 32
#define LDP (TK + 8)   // padded LDS row length in halfs (80B, keeps 16B alignment)

__global__ __launch_bounds__(256)
void gemm_bf16_wmma_kernel(const __bf16* __restrict__ A, const __bf16* __restrict__ Bt,
                           const float* __restrict__ bias, float* __restrict__ C,
                           int M, int K, int Nc)
{
    __shared__ __bf16 ldsA[2][TM][LDP];
    __shared__ __bf16 ldsB[2][TN][LDP];   // [out-col][k] (weights pre-transposed)

    const int t    = threadIdx.x;
    const int lane = t & 31;
    const int wave = t >> 5;             // 0..7
    const int m0   = (wave >> 1) * 32;   // wave row origin
    const int n0   = (wave & 1) * 64;    // wave col origin
    const int l15  = lane & 15;
    const int hi   = lane >> 4;          // 0 or 1

    const int blockM = blockIdx.y * TM;
    const int blockN = blockIdx.x * TN;

    v8f acc[2][4] = {};

    // cooperative tile load mapping (identical for A and Bt): 256 threads x 32B
    const int row = t >> 1;              // 0..127
    const int c16 = (t & 1) * 16;        // 0 or 16 halfs

    const __bf16* agBase = A  + (size_t)(blockM + row) * K + c16;
    const __bf16* bgBase = Bt + (size_t)(blockN + row) * K + c16;

    const unsigned laA0 = (unsigned)(size_t)&ldsA[0][row][c16];
    const unsigned laA1 = (unsigned)(size_t)&ldsA[1][row][c16];
    const unsigned laB0 = (unsigned)(size_t)&ldsB[0][row][c16];
    const unsigned laB1 = (unsigned)(size_t)&ldsB[1][row][c16];

    auto issue = [&](unsigned la, unsigned lb, int k0) {
        const __bf16* ag = agBase + k0;
        const __bf16* bg = bgBase + k0;
        asm volatile("global_load_async_to_lds_b128 %0, %1, off"
                     :: "v"(la), "v"((unsigned long long)(size_t)ag) : "memory");
        asm volatile("global_load_async_to_lds_b128 %0, %1, off"
                     :: "v"(la + 16u), "v"((unsigned long long)(size_t)(ag + 8)) : "memory");
        asm volatile("global_load_async_to_lds_b128 %0, %1, off"
                     :: "v"(lb), "v"((unsigned long long)(size_t)bg) : "memory");
        asm volatile("global_load_async_to_lds_b128 %0, %1, off"
                     :: "v"(lb + 16u), "v"((unsigned long long)(size_t)(bg + 8)) : "memory");
    };

    auto compute = [&](int s) {
        // A frag (16x32): lane<16 row=l15 K{0..7,16..23}; lane>=16 K{8..15,24..31}
        v16bf a0, a1;
        {
            const __bf16* p0 = &ldsA[s][m0 + l15][hi * 8];
            uint4* d0 = reinterpret_cast<uint4*>(&a0);
            d0[0] = *reinterpret_cast<const uint4*>(p0);
            d0[1] = *reinterpret_cast<const uint4*>(p0 + 16);
            const __bf16* p1 = &ldsA[s][m0 + 16 + l15][hi * 8];
            uint4* d1 = reinterpret_cast<uint4*>(&a1);
            d1[0] = *reinterpret_cast<const uint4*>(p1);
            d1[1] = *reinterpret_cast<const uint4*>(p1 + 16);
        }
        // B frags (32x16): lane<16 col=l15 K{0..7,16..23}; lane>=16 K{8..15,24..31}
        v16bf bf[4];
        #pragma unroll
        for (int j = 0; j < 4; ++j) {
            const __bf16* p = &ldsB[s][n0 + j * 16 + l15][hi * 8];
            uint4* d = reinterpret_cast<uint4*>(&bf[j]);
            d[0] = *reinterpret_cast<const uint4*>(p);
            d[1] = *reinterpret_cast<const uint4*>(p + 16);
        }
        #pragma unroll
        for (int j = 0; j < 4; ++j) {
            acc[0][j] = __builtin_amdgcn_wmma_f32_16x16x32_bf16(false, a0, false, bf[j], (short)0, acc[0][j], false, false);
            acc[1][j] = __builtin_amdgcn_wmma_f32_16x16x32_bf16(false, a1, false, bf[j], (short)0, acc[1][j], false, false);
        }
    };

    issue(laA0, laB0, 0);                       // prologue: tile 0 -> buffer 0

    // steady state: branch-free, always one tile in flight (K multiple of 2*TK, >= 4 tiles)
    int k0 = 0;
    for (; k0 + 2 * TK < K; k0 += 2 * TK) {
        issue(laA1, laB1, k0 + TK);
        asm volatile("s_wait_asynccnt 0x4" ::: "memory");  // tile for buf0 landed
        __syncthreads();
        compute(0);
        __syncthreads();                                   // buf0 free for reuse

        issue(laA0, laB0, k0 + 2 * TK);
        asm volatile("s_wait_asynccnt 0x4" ::: "memory");  // tile for buf1 landed
        __syncthreads();
        compute(1);
        __syncthreads();
    }

    // epilogue: k0 == K - 2*TK ; tiles k0 (in buf0, in flight) and k0+TK remain
    issue(laA1, laB1, k0 + TK);
    asm volatile("s_wait_asynccnt 0x4" ::: "memory");
    __syncthreads();
    compute(0);
    __syncthreads();

    asm volatile("s_wait_asynccnt 0x0" ::: "memory");
    __syncthreads();
    compute(1);

    // C/D layout: VGPR r -> row (lane<16 ? r : r+8), col = l15
    #pragma unroll
    for (int i = 0; i < 2; ++i) {
        const int gmBase = blockM + m0 + i * 16 + hi * 8;
        #pragma unroll
        for (int j = 0; j < 4; ++j) {
            const int gn = blockN + n0 + j * 16 + l15;
            const float bb = bias[gn];
            #pragma unroll
            for (int r = 0; r < 8; ++r) {
                C[(size_t)(gmBase + r) * Nc + gn] = acc[i][j][r] + bb;
            }
        }
    }
}

// ---------------- edge attention kernels ----------------

// one wave32 per (edge, head): s = scale * dot(q[row,h,:], k[col,h,:])
__global__ __launch_bounds__(256)
void edge_score_kernel(const float* __restrict__ q, const float* __restrict__ k,
                       const int* __restrict__ edges, float* __restrict__ scores,
                       unsigned* __restrict__ maxenc, float scale)
{
    const int wid  = (int)((blockIdx.x * blockDim.x + threadIdx.x) >> 5);
    const int lane = threadIdx.x & 31;
    if (wid >= EE * HH) return;
    const int e = wid >> 2;
    const int h = wid & 3;
    const int r = edges[2 * e], c = edges[2 * e + 1];
    const float4* qp = reinterpret_cast<const float4*>(q + (size_t)r * DD + h * HD + lane * 16);
    const float4* kp = reinterpret_cast<const float4*>(k + (size_t)c * DD + h * HD + lane * 16);
    float s = 0.f;
    #pragma unroll
    for (int i = 0; i < 4; ++i) {
        float4 qa = qp[i], ka = kp[i];
        s += qa.x * ka.x + qa.y * ka.y + qa.z * ka.z + qa.w * ka.w;
    }
    #pragma unroll
    for (int off = 16; off > 0; off >>= 1) s += __shfl_xor(s, off, 32);
    if (lane == 0) {
        s *= scale;
        scores[wid] = s;
        atomicMax(&maxenc[r * HH + h], f2ord(s));
    }
}

// one thread per edge: first occurrence of (row,col) adds exp(s-max) to denom (mask dedup)
__global__ void edge_denom_kernel(const int* __restrict__ edges, const float* __restrict__ scores,
                                  const unsigned* __restrict__ maxenc, float* __restrict__ denom,
                                  unsigned* __restrict__ dedup)
{
    const int e = blockIdx.x * blockDim.x + threadIdx.x;
    if (e >= EE) return;
    const int r = edges[2 * e], c = edges[2 * e + 1];
    const unsigned long long key = (unsigned long long)r * (unsigned long long)NN + (unsigned long long)c;
    const unsigned word = (unsigned)(key >> 5);
    const unsigned bit  = 1u << (key & 31u);
    const unsigned old  = atomicOr(&dedup[word], bit);
    if (old & bit) return;   // duplicate edge: denominator counts it once
    #pragma unroll
    for (int h = 0; h < HH; ++h) {
        const float mx = ord2f(maxenc[r * HH + h]);
        atomicAdd(&denom[r * HH + h], expf(scores[e * HH + h] - mx));
    }
}

// one wave32 per (edge, head): agg[row] += (exp(s-max)/denom) * v[col]  (per-edge, duplicates count)
__global__ __launch_bounds__(256)
void edge_agg_kernel(const int* __restrict__ edges, const float* __restrict__ scores,
                     const unsigned* __restrict__ maxenc, const float* __restrict__ denom,
                     const float* __restrict__ v, float* __restrict__ agg)
{
    const int wid  = (int)((blockIdx.x * blockDim.x + threadIdx.x) >> 5);
    const int lane = threadIdx.x & 31;
    if (wid >= EE * HH) return;
    const int e = wid >> 2;
    const int h = wid & 3;
    const int r = edges[2 * e], c = edges[2 * e + 1];
    const float mx = ord2f(maxenc[r * HH + h]);
    const float w  = expf(scores[e * HH + h] - mx) / denom[r * HH + h];
    const float* vp = v + (size_t)c * DD + h * HD + lane * 16;
    float* ap       = agg + (size_t)r * DD + h * HD + lane * 16;
    #pragma unroll
    for (int i = 0; i < 16; ++i) atomicAdd(&ap[i], w * vp[i]);
}

// ---------------- host launch ----------------

extern "C" void kernel_launch(void* const* d_in, const int* in_sizes, int n_in,
                              void* d_out, int out_size, void* d_ws, size_t ws_size,
                              hipStream_t stream) {
    const float* feats = (const float*)d_in[0];
    const float* Wq    = (const float*)d_in[1];
    const float* bq    = (const float*)d_in[2];
    const float* Wk    = (const float*)d_in[3];
    const float* bk    = (const float*)d_in[4];
    const float* Wv    = (const float*)d_in[5];
    const float* bv    = (const float*)d_in[6];
    const float* Wo    = (const float*)d_in[7];
    const float* bo    = (const float*)d_in[8];
    const int*   edges = (const int*)d_in[9];
    float* out = (float*)d_out;

    // --- workspace layout (bytes) ---
    constexpr size_t WB   = (size_t)LL * DD * DD * 2;   // one weight set (TRANSPOSED), bf16
    constexpr size_t XB   = (size_t)NN * DD * 2;        // bf16 activations
    constexpr size_t XF   = (size_t)NN * DD * 4;        // fp32 activations
    constexpr size_t SCB  = (size_t)EE * HH * 4;
    constexpr size_t STB  = (size_t)NN * HH * 4;
    constexpr size_t DEDB = (size_t)NN * NN / 8;

    char* p = (char*)d_ws;
    __bf16* Wqb  = (__bf16*)p;            p += WB;
    __bf16* Wkb  = (__bf16*)p;            p += WB;
    __bf16* Wvb  = (__bf16*)p;            p += WB;
    __bf16* Wob  = (__bf16*)p;            p += WB;
    __bf16* xb   = (__bf16*)p;            p += XB;
    __bf16* aggb = (__bf16*)p;            p += XB;
    float*  qbuf = (float*)p;             p += XF;
    float*  kbuf = (float*)p;             p += XF;
    float*  vbuf = (float*)p;             p += XF;
    float*  xtmp = (float*)p;             p += XF;
    float*  scr  = (float*)p;             p += SCB;
    // contiguous zero-region: agg | dedup | maxenc | denom
    char*   zreg = p;
    float*    agg    = (float*)p;         p += XF;
    unsigned* dedup  = (unsigned*)p;      p += DEDB;
    unsigned* maxenc = (unsigned*)p;      p += STB;
    float*    denom  = (float*)p;         p += STB;
    const size_t zbytes = (size_t)(p - zreg);

    const dim3 gemmGrid(DD / TN, NN / TM);   // (16, 32)
    const dim3 tpGrid(DD / 32, DD / 32, LL); // weight transpose tiles
    const float scale = 1.0f / sqrtf((float)HD);

    // weights: transpose + convert to bf16 (Bt[j][k] = W[k][j]) so GEMM B-tiles are K-contiguous
    transpose_f32_to_bf16_kernel<<<tpGrid, 256, 0, stream>>>(Wq, Wqb);
    transpose_f32_to_bf16_kernel<<<tpGrid, 256, 0, stream>>>(Wk, Wkb);
    transpose_f32_to_bf16_kernel<<<tpGrid, 256, 0, stream>>>(Wv, Wvb);
    transpose_f32_to_bf16_kernel<<<tpGrid, 256, 0, stream>>>(Wo, Wob);
    f32_to_bf16_kernel<<<4096, 256, 0, stream>>>(feats, xb, (size_t)NN * DD);

    for (int l = 0; l < LL; ++l) {
        // zero agg/dedup/max/denom
        zero16_kernel<<<2048, 256, 0, stream>>>((uint4*)zreg, zbytes / 16);

        const size_t wOff = (size_t)l * DD * DD;
        gemm_bf16_wmma_kernel<<<gemmGrid, 256, 0, stream>>>(xb, Wqb + wOff, bq + (size_t)l * DD, qbuf, NN, DD, DD);
        gemm_bf16_wmma_kernel<<<gemmGrid, 256, 0, stream>>>(xb, Wkb + wOff, bk + (size_t)l * DD, kbuf, NN, DD, DD);
        gemm_bf16_wmma_kernel<<<gemmGrid, 256, 0, stream>>>(xb, Wvb + wOff, bv + (size_t)l * DD, vbuf, NN, DD, DD);

        // attention over edges
        const int ehWaves  = EE * HH;                       // 524288 waves
        const int ehBlocks = (ehWaves * 32) / 256;          // 65536 blocks
        edge_score_kernel<<<ehBlocks, 256, 0, stream>>>(qbuf, kbuf, edges, scr, maxenc, scale);
        edge_denom_kernel<<<EE / 256, 256, 0, stream>>>(edges, scr, maxenc, denom, dedup);
        edge_agg_kernel<<<ehBlocks, 256, 0, stream>>>(edges, scr, maxenc, denom, vbuf, agg);

        // output projection
        f32_to_bf16_kernel<<<4096, 256, 0, stream>>>(agg, aggb, (size_t)NN * DD);
        float* dst = (l == LL - 1) ? out : xtmp;
        gemm_bf16_wmma_kernel<<<gemmGrid, 256, 0, stream>>>(aggb, Wob + wOff, bo + (size_t)l * DD, dst, NN, DD, DD);

        if (l < LL - 1)
            f32_to_bf16_kernel<<<4096, 256, 0, stream>>>(xtmp, xb, (size_t)NN * DD);
    }
}
